// SelfAttentionBlock_35905926594931
// MI455X (gfx1250) — matrix operand order repaired and checked by reading, compiled, not run on hardware
//
#include <hip/hip_runtime.h>
#include <hip/hip_bf16.h>
#include <stdint.h>

#define H_DIM 1024
#define SEQ   2048
#define NHEAD 16
#define HDIM  64

typedef __attribute__((ext_vector_type(16))) __bf16 v16bf;
typedef __attribute__((ext_vector_type(2)))  __bf16 v2bf;
typedef __attribute__((ext_vector_type(8)))  float  v8f;

struct Frag32 { uint4 lo, hi; };   // 32 bytes == v16bf

static __device__ __forceinline__ v16bf frag_cast(const Frag32& f) {
  return __builtin_bit_cast(v16bf, f);
}
static __device__ __forceinline__ unsigned short f2bf(float x) {
  unsigned u = __builtin_bit_cast(unsigned, x);
  u += 0x7FFFu + ((u >> 16) & 1u);            // round-to-nearest-even
  return (unsigned short)(u >> 16);
}
// Packed fp32x2 -> bf16x2 (single v_cvt_pk_bf16_f32 when available)
static __device__ __forceinline__ unsigned pack2bf(float lo, float hi) {
#if __has_builtin(__builtin_amdgcn_cvt_pk_bf16_f32)
  v2bf p = __builtin_amdgcn_cvt_pk_bf16_f32(lo, hi);
  return __builtin_bit_cast(unsigned, p);
#else
  return (unsigned)f2bf(lo) | ((unsigned)f2bf(hi) << 16);
#endif
}
static __device__ __forceinline__ unsigned short f2bf_s(float x) {
#if __has_builtin(__builtin_amdgcn_cvt_pk_bf16_f32)
  return (unsigned short)(pack2bf(x, 0.0f) & 0xFFFFu);
#else
  return f2bf(x);
#endif
}
static __device__ __forceinline__ v8f vzero() {
  v8f z;
  #pragma unroll
  for (int i = 0; i < 8; ++i) z[i] = 0.0f;
  return z;
}
static __device__ __forceinline__ v8f wmma_bf16(const Frag32& a, const Frag32& b, v8f c) {
  return __builtin_amdgcn_wmma_f32_16x16x32_bf16(
      /*neg_a=*/false, frag_cast(a), /*neg_b=*/false, frag_cast(b),
      /*c_mod=*/(short)0, c, /*reuse_a=*/false, /*reuse_b=*/false);
}

// Async global->LDS copy of one 16-byte chunk (tracked by ASYNCcnt).
static __device__ __forceinline__ void async_copy_b128(unsigned lds_addr,
                                                       unsigned goff,
                                                       const void* gbase) {
  asm volatile("global_load_async_to_lds_b128 %0, %1, %2"
               :: "v"(lds_addr), "v"(goff), "s"(gbase)
               : "memory");
}
static __device__ __forceinline__ void wait_async0() {
#if __has_builtin(__builtin_amdgcn_s_wait_asynccnt)
  __builtin_amdgcn_s_wait_asynccnt(0);
#else
  asm volatile("s_wait_asynccnt 0x0" ::: "memory");
#endif
}

// 2 * log2(e): temperature (1/0.5) and exp->exp2 conversion folded into Q.
#define QSCALE 2.8853900817779268f

// ---------------------------------------------------------------------------
// Tiled GEMM: Y[m,n] = sum_k A[m,k] * W[n,k] + bias[n]
// MODE 0: A = feat fp32, blockIdx.z selects {q,k,v}; output bf16
//         z==0 (Q): head-major, scaled by 2*log2e;  z==1 (K): head-major;
//         z==2 (V): transposed [B*NH][HD][S]
// MODE 1: A = attn bf16; output fp32 with bias+ReLU into OutF
// Block: 256 threads (8 waves), tile 128x256, wave tile 64x64, BK=32.
// Double-buffered LDS, one barrier per K-step.
// ---------------------------------------------------------------------------
template<int MODE>
__global__ __launch_bounds__(256) void gemm_kernel(
    const float* Af, const unsigned short* Ah,
    const float* W0, const float* W1, const float* W2,
    const float* B0, const float* B1, const float* B2,
    unsigned short* O0, unsigned short* O1, unsigned short* O2,
    float* OutF)
{
  constexpr int BM = 128, BN = 256, BK = 32, LDT = 40;
  __shared__ unsigned short sA[2][BM * LDT];
  __shared__ unsigned short sB[2][BN * LDT];

  const int tid  = threadIdx.x;
  const int lane = tid & 31;
  const int wid  = tid >> 5;
  const int g    = (lane >> 4) & 1;
  const int l16  = lane & 15;
  const int wm   = wid >> 2;      // 0..1 -> 64 rows each
  const int wn   = wid & 3;       // 0..3 -> 64 cols each
  const int m0   = blockIdx.y * BM;
  const int n0   = blockIdx.x * BN;

  const float* W = W0; const float* bias = B0; unsigned short* Ob = O0;
  if (MODE == 0) {
    if      (blockIdx.z == 1) { W = W1; bias = B1; Ob = O1; }
    else if (blockIdx.z == 2) { W = W2; bias = B2; Ob = O2; }
  }

  v8f acc[4][4];
  #pragma unroll
  for (int i = 0; i < 4; ++i)
    #pragma unroll
    for (int j = 0; j < 4; ++j) acc[i][j] = vzero();

  const int lrow = tid >> 3;        // 0..31
  const int lcol = (tid & 7) * 4;   // 0..28

  float4 ra[4]; uint2 rha[4]; float4 rb[8];

  auto gload = [&](int kb) {        // global tile -> registers (prefetch)
    #pragma unroll
    for (int p = 0; p < 4; ++p) {
      int r = p * 32 + lrow;
      if (MODE == 0) ra[p]  = *(const float4*)(Af + (size_t)(m0 + r) * H_DIM + kb * BK + lcol);
      else           rha[p] = *(const uint2*)(Ah + (size_t)(m0 + r) * H_DIM + kb * BK + lcol);
    }
    #pragma unroll
    for (int p = 0; p < 8; ++p) {
      int r = p * 32 + lrow;
      rb[p] = *(const float4*)(W + (size_t)(n0 + r) * H_DIM + kb * BK + lcol);
    }
  };
  auto sstore = [&](int buf) {      // registers -> LDS (convert to bf16)
    #pragma unroll
    for (int p = 0; p < 4; ++p) {
      int r = p * 32 + lrow;
      if (MODE == 0) {
        uint2 u;
        u.x = pack2bf(ra[p].x, ra[p].y);
        u.y = pack2bf(ra[p].z, ra[p].w);
        *(uint2*)&sA[buf][r * LDT + lcol] = u;
      } else {
        *(uint2*)&sA[buf][r * LDT + lcol] = rha[p];
      }
    }
    #pragma unroll
    for (int p = 0; p < 8; ++p) {
      int r = p * 32 + lrow;
      uint2 u;
      u.x = pack2bf(rb[p].x, rb[p].y);
      u.y = pack2bf(rb[p].z, rb[p].w);
      *(uint2*)&sB[buf][r * LDT + lcol] = u;
    }
  };

  gload(0);
  sstore(0);
  __syncthreads();

  for (int kb = 0; kb < H_DIM / BK; ++kb) {
    const int cur = kb & 1;
    const bool more = (kb + 1) < (H_DIM / BK);
    if (more) gload(kb + 1);        // issue early: LOADcnt wait lands after WMMAs

    Frag32 fa[4], fb[4];
    #pragma unroll
    for (int i = 0; i < 4; ++i) {
      const unsigned short* bp = &sA[cur][(wm * 64 + i * 16 + l16) * LDT];
      fa[i].lo = *(const uint4*)(bp + 8 * g);        // e0..7 : K = 8g+e
      fa[i].hi = *(const uint4*)(bp + 16 + 8 * g);   // e8..15: K = 16+8g+e
    }
    #pragma unroll
    for (int j = 0; j < 4; ++j) {
      const unsigned short* bp = &sB[cur][(wn * 64 + j * 16 + l16) * LDT];
      fb[j].lo = *(const uint4*)(bp + 8 * g);        // e0..7 : K = 8g+e
      fb[j].hi = *(const uint4*)(bp + 16 + 8 * g);   // e8..15: K = 16+8g+e
    }
    #pragma unroll
    for (int i = 0; i < 4; ++i)
      #pragma unroll
      for (int j = 0; j < 4; ++j)
        acc[i][j] = wmma_bf16(fa[i], fb[j], acc[i][j]);

    if (more) sstore(cur ^ 1);
    __syncthreads();
  }

  // epilogue (C layout: row = r + 8g, col = l16)
  #pragma unroll
  for (int i = 0; i < 4; ++i) {
    #pragma unroll
    for (int j = 0; j < 4; ++j) {
      int n = n0 + wn * 64 + j * 16 + l16;
      float bv = bias[n];
      #pragma unroll
      for (int r = 0; r < 8; ++r) {
        int m = m0 + wm * 64 + i * 16 + r + 8 * g;
        float v = acc[i][j][r] + bv;
        if (MODE == 0) {
          int b = m >> 11, s = m & 2047;
          int h = n >> 6,  d = n & 63;
          size_t idx;
          if (blockIdx.z == 2)
            idx = ((size_t)(b * NHEAD + h) * HDIM + d) * SEQ + s;   // V transposed
          else
            idx = ((size_t)(b * NHEAD + h) * SEQ + s) * HDIM + d;
          if (blockIdx.z == 0) v *= QSCALE;                          // fold temp+log2e into Q
          Ob[idx] = f2bf_s(v);
        } else {
          OutF[(size_t)m * H_DIM + n] = fmaxf(v, 0.0f);
        }
      }
    }
  }
}

// ---------------------------------------------------------------------------
// Flash attention: grid (S/64 query tiles, B*NH heads), 128 threads (4 waves),
// each wave owns 16 query rows. K ([seq][dim]) and Vt ([dim][seq]) tiles are
// double-buffered in LDS via GLOBAL_LOAD_ASYNC_TO_LDS_B128 (ASYNCcnt).
// Softmax in log2 domain; row-sum computed on the matrix pipe (P @ ones).
// ---------------------------------------------------------------------------
__global__ __launch_bounds__(128) void attn_kernel(
    const unsigned short* Q, const unsigned short* K, const unsigned short* Vt,
    unsigned short* AO)
{
  constexpr int LDK = 72;                       // 144B row stride (16B aligned)
  __shared__ unsigned short sK[2][64 * LDK];    // [key][dim]
  __shared__ unsigned short sV[2][64 * LDK];    // [dim][key] (already transposed)
  __shared__ unsigned short sP[4 * 16 * LDK];   // per-wave P tile [row][key]

  const int tid  = threadIdx.x;
  const int lane = tid & 31;
  const int wid  = tid >> 5;
  const int g    = (lane >> 4) & 1;
  const int l16  = lane & 15;
  const int head = blockIdx.y;
  const int qt   = blockIdx.x;

  const unsigned short* Qh = Q  + (size_t)head * SEQ * HDIM;   // [seq][dim], pre-scaled
  const unsigned short* Kh = K  + (size_t)head * SEQ * HDIM;   // [seq][dim]
  const unsigned short* Vh = Vt + (size_t)head * HDIM * SEQ;   // [dim][seq]

  // Q fragments (A layout), resident for the whole pass
  const int qrow = qt * 64 + wid * 16 + l16;
  Frag32 fq[2];
  #pragma unroll
  for (int c = 0; c < 2; ++c) {
    const unsigned short* bp = Qh + (size_t)qrow * HDIM + c * 32;
    fq[c].lo = *(const uint4*)(bp + 8 * g);
    fq[c].hi = *(const uint4*)(bp + 16 + 8 * g);
  }

  // Constant all-ones B fragment (bf16 1.0) for row-sum via WMMA.
  Frag32 fones;
  {
    uint4 ones; ones.x = ones.y = ones.z = ones.w = 0x3F803F80u;
    fones.lo = ones; fones.hi = ones;
  }

  float mstat[8], lstat[8];
  #pragma unroll
  for (int r = 0; r < 8; ++r) { mstat[r] = -1e30f; lstat[r] = 0.0f; }
  v8f o[4];
  #pragma unroll
  for (int j = 0; j < 4; ++j) o[j] = vzero();

  // Issue the async copies for tile t into buffer buf.
  // 64 rows x 128B per matrix = 512 b128 chunks -> 4 per thread per matrix.
  auto stage = [&](int t, int buf) {
    #pragma unroll
    for (int i = 0; i < 4; ++i) {
      int chunk = i * 128 + tid;          // 0..511, consecutive tid = consecutive chunks
      int row   = chunk >> 3;             // 0..63 (key for K, dim for Vt)
      int co    = (chunk & 7) * 16;       // byte offset inside a 128B row
      unsigned kgoff = (unsigned)((t * 64 + row) * (HDIM * 2) + co);
      unsigned klds  = (unsigned)(uintptr_t)(&sK[buf][row * LDK]) + (unsigned)co;
      async_copy_b128(klds, kgoff, (const void*)Kh);
      unsigned vgoff = (unsigned)(row * (SEQ * 2) + t * 64 * 2 + co);
      unsigned vlds  = (unsigned)(uintptr_t)(&sV[buf][row * LDK]) + (unsigned)co;
      async_copy_b128(vlds, vgoff, (const void*)Vh);
    }
  };

  stage(0, 0);

  for (int kt = 0; kt < SEQ / 64; ++kt) {
    const int buf = kt & 1;
    wait_async0();          // this wave's copies for tile kt are in LDS
    __syncthreads();        // everyone's copies done; prev compute done
    if (kt + 1 < SEQ / 64) stage(kt + 1, buf ^ 1);   // prefetch next tile

    const unsigned short* cK = &sK[buf][0];
    const unsigned short* cV = &sV[buf][0];

    // scores (log2 domain) = Qs @ K^T  (B-frag: lane = key, stripe head-dim)
    v8f sc[4];
    #pragma unroll
    for (int j = 0; j < 4; ++j) {
      sc[j] = vzero();
      #pragma unroll
      for (int c = 0; c < 2; ++c) {
        Frag32 fk;
        const unsigned short* bp = &cK[(j * 16 + l16) * LDK + c * 32];
        fk.lo = *(const uint4*)(bp + 8 * g);
        fk.hi = *(const uint4*)(bp + 16 + 8 * g);
        sc[j] = wmma_bf16(fq[c], fk, sc[j]);
      }
    }

    // online softmax (exp2): row max via shuffle tree over the 16-lane half
    float rm[8];
    #pragma unroll
    for (int r = 0; r < 8; ++r) {
      float v = fmaxf(fmaxf(sc[0][r], sc[1][r]), fmaxf(sc[2][r], sc[3][r]));
      #pragma unroll
      for (int msk = 1; msk < 16; msk <<= 1) v = fmaxf(v, __shfl_xor(v, msk, 16));
      rm[r] = v;
    }
    float alpha[8];
    #pragma unroll
    for (int r = 0; r < 8; ++r) {
      float mn = fmaxf(mstat[r], rm[r]);
      alpha[r] = exp2f(mstat[r] - mn);
      mstat[r] = mn;
    }

    // p = exp2(s - m): convert + park in LDS (C layout -> row-major)
    #pragma unroll
    for (int j = 0; j < 4; ++j)
      #pragma unroll
      for (int r = 0; r < 8; ++r)
        sP[(wid * 16 + r + 8 * g) * LDK + j * 16 + l16] =
            f2bf_s(exp2f(sc[j][r] - mstat[r]));

    #pragma unroll
    for (int j = 0; j < 4; ++j)
      #pragma unroll
      for (int r = 0; r < 8; ++r) o[j][r] *= alpha[r];
    __syncthreads();

    // P as A-layout fragments
    Frag32 fp[2];
    #pragma unroll
    for (int c = 0; c < 2; ++c) {
      const unsigned short* bp = &sP[(wid * 16 + l16) * LDK + c * 32];
      fp[c].lo = *(const uint4*)(bp + 8 * g);
      fp[c].hi = *(const uint4*)(bp + 16 + 8 * g);
    }

    // row-sum on the matrix pipe: P @ ones -> every lane of the half holds
    // its row's sum (C layout replicates across the 16 columns)
    v8f rsum = vzero();
    rsum = wmma_bf16(fp[0], fones, rsum);
    rsum = wmma_bf16(fp[1], fones, rsum);
    #pragma unroll
    for (int r = 0; r < 8; ++r) lstat[r] = lstat[r] * alpha[r] + rsum[r];

    // O += P @ V  (B-frag: lane = head-dim, stripe keys from transposed sV)
    #pragma unroll
    for (int c = 0; c < 2; ++c)
      #pragma unroll
      for (int j = 0; j < 4; ++j) {
        Frag32 fv;
        const unsigned short* bp = &cV[(j * 16 + l16) * LDK + c * 32];
        fv.lo = *(const uint4*)(bp + 8 * g);
        fv.hi = *(const uint4*)(bp + 16 + 8 * g);
        o[j] = wmma_bf16(fp[c], fv, o[j]);
      }
  }

  // write attention output as [B*S][H] bf16 for the output projection
  float inv[8];
  #pragma unroll
  for (int r = 0; r < 8; ++r) inv[r] = 1.0f / lstat[r];
  const int b = head >> 4, h = head & 15;
  #pragma unroll
  for (int j = 0; j < 4; ++j) {
    int col = h * HDIM + j * 16 + l16;
    #pragma unroll
    for (int r = 0; r < 8; ++r) {
      int s = qt * 64 + wid * 16 + r + 8 * g;
      float v = o[j][r] * inv[r];
      AO[((size_t)b * SEQ + s) * H_DIM + col] = f2bf_s(v);
    }
  }
}

// ---------------------------------------------------------------------------
// LayerNorm, in-place on d_out: one 256-thread block per row of 1024
// ---------------------------------------------------------------------------
__global__ __launch_bounds__(256) void ln_kernel(float* Y, const float* gw, const float* bw)
{
  __shared__ float red[256];
  const int tid = threadIdx.x;
  float* y = Y + (size_t)blockIdx.x * H_DIM;
  float v[4];
  float s = 0.0f;
  #pragma unroll
  for (int i = 0; i < 4; ++i) { v[i] = y[tid + i * 256]; s += v[i]; }
  red[tid] = s; __syncthreads();
  for (int off = 128; off > 0; off >>= 1) {
    if (tid < off) red[tid] += red[tid + off];
    __syncthreads();
  }
  float mu = red[0] * (1.0f / H_DIM);
  __syncthreads();
  s = 0.0f;
  #pragma unroll
  for (int i = 0; i < 4; ++i) { float d = v[i] - mu; s += d * d; }
  red[tid] = s; __syncthreads();
  for (int off = 128; off > 0; off >>= 1) {
    if (tid < off) red[tid] += red[tid + off];
    __syncthreads();
  }
  float inv = rsqrtf(red[0] * (1.0f / H_DIM) + 1e-5f);
  #pragma unroll
  for (int i = 0; i < 4; ++i) {
    int c = tid + i * 256;
    y[c] = (v[i] - mu) * inv * gw[c] + bw[c];
  }
}

// ---------------------------------------------------------------------------
extern "C" void kernel_launch(void* const* d_in, const int* in_sizes, int n_in,
                              void* d_out, int out_size, void* d_ws, size_t ws_size,
                              hipStream_t stream)
{
  const float* feat  = (const float*)d_in[0];
  const float* qw    = (const float*)d_in[1];
  const float* qb    = (const float*)d_in[2];
  const float* kw    = (const float*)d_in[3];
  const float* kb    = (const float*)d_in[4];
  const float* vw    = (const float*)d_in[5];
  const float* vb    = (const float*)d_in[6];
  const float* ow    = (const float*)d_in[7];
  const float* obias = (const float*)d_in[8];
  const float* ln_g  = (const float*)d_in[9];
  const float* ln_b  = (const float*)d_in[10];
  float* out = (float*)d_out;

  const size_t TOK = (size_t)4 * SEQ;                 // 8192 rows
  unsigned short* qws = (unsigned short*)d_ws;        // bf16 [B*NH][S][HD], pre-scaled
  unsigned short* kws = qws + TOK * H_DIM;            // bf16 [B*NH][S][HD]
  unsigned short* vws = kws + TOK * H_DIM;            // bf16 [B*NH][HD][S] (transposed)
  unsigned short* aws = vws + TOK * H_DIM;            // bf16 [B*S][H]

  dim3 g1(H_DIM / 256, (unsigned)(TOK / 128), 3);
  gemm_kernel<0><<<g1, 256, 0, stream>>>(feat, nullptr, qw, kw, vw, qb, kb, vb,
                                         qws, kws, vws, nullptr);

  attn_kernel<<<dim3(SEQ / 64, 4 * NHEAD), 128, 0, stream>>>(qws, kws, vws, aws);

  dim3 g2(H_DIM / 256, (unsigned)(TOK / 128), 1);
  gemm_kernel<1><<<g2, 256, 0, stream>>>(nullptr, aws, ow, ow, ow, obias, obias, obias,
                                         nullptr, nullptr, nullptr, out);

  ln_kernel<<<(unsigned)TOK, 256, 0, stream>>>(out, ln_g, ln_b);
}
